// SoftMaskedAttention_89644557402757
// MI455X (gfx1250) — matrix operand order repaired and checked by reading, not compiled, not run on hardware
//
#include <hip/hip_runtime.h>
#include <hip/hip_bf16.h>

// ---------------------------------------------------------------------------
// SoftMaskedAttention on MI455X (gfx1250): all-WMMA pipeline.
//   Stage 1: Q/K/V projections (f32 -> f16 head-major; V stored transposed;
//            Q pre-scaled by (1/sqrt(D))*log2(e) for log2-domain softmax)
//   Stage 2: split-S flash attention (4 partitions) -> unnormalized partials
//   Stage 3: partial combine (log2-domain rescale + normalize)
//   Stage 4: output projection -> f32
// ---------------------------------------------------------------------------

typedef __attribute__((ext_vector_type(16))) _Float16 v16h;
typedef __attribute__((ext_vector_type(8)))  _Float16 v8h;
typedef __attribute__((ext_vector_type(8)))  float    v8f;
typedef __attribute__((ext_vector_type(4)))  float    v4f;

#define WMMA_F16(a, b, c) \
  __builtin_amdgcn_wmma_f32_16x16x32_f16(false, (a), false, (b), (short)0, (c), false, false)

#define LOG2E 1.44269504088896340f

// ---- Fragment loaders -----------------------------------------------------
// A fragment (16x32, M x K) from row-major [16, >=32] f32 source.
// ISA: lanes 0-15 = rows M, K 0..7 (v0-3) + 16..23 (v4-7); lanes 16-31: K 8..15 + 24..31.
__device__ __forceinline__ v16h load_a_f32(const float* __restrict__ p0, int ld) {
  int lane = threadIdx.x & 31;
  const float* p = p0 + (size_t)(lane & 15) * (size_t)ld + ((lane >> 4) << 3);
  v4f x0 = *(const v4f*)(p);
  v4f x1 = *(const v4f*)(p + 4);
  v4f x2 = *(const v4f*)(p + 16);
  v4f x3 = *(const v4f*)(p + 20);
  v16h f;
#pragma unroll
  for (int j = 0; j < 4; ++j) {
    f[j]      = (_Float16)x0[j];
    f[4 + j]  = (_Float16)x1[j];
    f[8 + j]  = (_Float16)x2[j];
    f[12 + j] = (_Float16)x3[j];
  }
  return f;
}

// B fragment (32x16, K x N) from row-major N x K (i.e. B^T) f32 source.
// ISA: lanes 0-15 = cols N, K 0..7 + 16..23; lanes 16-31: K 8..15 + 24..31.
__device__ __forceinline__ v16h load_b_f32(const float* __restrict__ p0, int ld) {
  int lane = threadIdx.x & 31;
  const float* p = p0 + (size_t)(lane & 15) * (size_t)ld + ((lane >> 4) << 3);
  v16h f;
#pragma unroll
  for (int q = 0; q < 4; ++q) {
    v4f x = *(const v4f*)(p + 4 * (q & 1) + 16 * (q >> 1));
#pragma unroll
    for (int j = 0; j < 4; ++j) f[4 * q + j] = (_Float16)x[j];
  }
  return f;
}

// A fragment from row-major [16, >=32] f16 source (global or LDS).
__device__ __forceinline__ v16h load_a_f16(const _Float16* __restrict__ p0, int ld) {
  int lane = threadIdx.x & 31;
  const _Float16* p = p0 + (size_t)(lane & 15) * (size_t)ld + ((lane >> 4) << 3);
  v8h lo = *(const v8h*)(p);
  v8h hi = *(const v8h*)(p + 16);
  v16h f;
#pragma unroll
  for (int j = 0; j < 8; ++j) { f[j] = lo[j]; f[8 + j] = hi[j]; }
  return f;
}

// B fragment from row-major N x K f16 source (each lane: K 8h..8h+7 and 16+8h..16+8h+7).
__device__ __forceinline__ v16h load_b_f16(const _Float16* __restrict__ p0, int ld) {
  int lane = threadIdx.x & 31;
  const _Float16* p = p0 + (size_t)(lane & 15) * (size_t)ld + ((lane >> 4) << 3);
  v8h lo = *(const v8h*)(p);
  v8h hi = *(const v8h*)(p + 16);
  v16h f;
#pragma unroll
  for (int j = 0; j < 8; ++j) { f[j] = lo[j]; f[8 + j] = hi[j]; }
  return f;
}

// ---- Stage 1: input projections -------------------------------------------
// out(n,e) = (X @ W^T + bias)(n,e) * scale, scattered to per-head f16 layout.
// transposed==0: dst[((b*8+h)*rpb + t)*32 + d]   (Q / K)
// transposed==1: dst[((b*8+h)*32 + d)*rpb + t]   (V^T)
__global__ void __launch_bounds__(128)
proj_kernel(const float* __restrict__ X, const float* __restrict__ W,
            const float* __restrict__ bias, _Float16* __restrict__ dst,
            int rpb, float scale, int transposed) {
  int wave  = threadIdx.x >> 5;
  int tile  = blockIdx.x * 4 + wave;
  int mtile = tile >> 4;   // 256/16 = 16 e-tiles per m-tile row
  int etile = tile & 15;
  int lane  = threadIdx.x & 31;

  const float* A  = X + (size_t)mtile * 16 * 256;
  const float* Bp = W + (size_t)etile * 16 * 256;
  v8f c = {};
#pragma unroll
  for (int k0 = 0; k0 < 256; k0 += 32) {
    v16h a = load_a_f32(A + k0, 256);
    v16h b = load_b_f32(Bp + k0, 256);
    c = WMMA_F16(a, b, c);
  }
  int half = lane >> 4, nn = lane & 15;
  int e = etile * 16 + nn;
  int h = e >> 5, d = e & 31;
  float bval = bias[e];
#pragma unroll
  for (int r = 0; r < 8; ++r) {
    int m  = mtile * 16 + r + 8 * half;   // global row in X
    int b_ = m / rpb;
    int t  = m % rpb;
    float val = (c[r] + bval) * scale;
    size_t idx = transposed
        ? ((((size_t)b_ * 8 + h) * 32 + d) * (size_t)rpb + t)
        : ((((size_t)b_ * 8 + h) * (size_t)rpb + t) * 32 + d);
    dst[idx] = (_Float16)val;
  }
}

// ---- Stage 2: split-S flash attention -------------------------------------
// Grid: (B*H) x NSPLIT x (TTILES/4) blocks, 4 waves each; every wave owns one
// 16-row t-tile over one S partition of 4096 keys, streamed in chunks of 64.
// Scores/bias are in log2 domain (log2e folded into Q scale and sf[h]).
#define NSPLIT 4
__global__ void __launch_bounds__(128)
fa_kernel(const _Float16* __restrict__ qh, const _Float16* __restrict__ kh,
          const _Float16* __restrict__ vt, const float* __restrict__ mask,
          const float* __restrict__ sf, float* __restrict__ po,
          float* __restrict__ pml) {
  const int S = 16384, T = 256, D = 32;
  __shared__ __align__(16) _Float16 lds_p[4][16 * 64];  // 2 KB per wave

  int bi = blockIdx.x;
  int tgroup = bi & 3;
  int split  = (bi >> 2) & (NSPLIT - 1);
  int bh     = bi >> 4;                  // b*8 + h
  int b  = bh >> 3, h = bh & 7;
  int wave = threadIdx.x >> 5;
  int lane = threadIdx.x & 31;
  int half = lane >> 4, nn = lane & 15;
  int ttile = tgroup * 4 + wave;
  int t0 = ttile * 16;
  int s_begin = split * (S / NSPLIT);
  int s_end   = s_begin + (S / NSPLIT);

  const _Float16* qbase = qh + ((size_t)bh * T + t0) * D;
  const _Float16* kbase = kh + (size_t)bh * S * D;
  const _Float16* vbase = vt + (size_t)bh * D * S;
  const float*    mbase = mask + ((size_t)b * T + t0) * S;
  float sfh = sf[h] * LOG2E;             // bias into log2 domain

  v16h aq = load_a_f16(qbase, D);        // Q pre-scaled by (1/sqrt(D))*log2e
  float mrun[8], lrun[8];
#pragma unroll
  for (int r = 0; r < 8; ++r) { mrun[r] = -1.0e30f; lrun[r] = 0.0f; }
  v8f o0 = {}, o1 = {};                  // O accumulators, d=0..15 / 16..31
  _Float16* pl = &lds_p[wave][0];

  for (int s0 = s_begin; s0 < s_end; s0 += 64) {
    // ---- scores: four 16x16 tiles over K-dim D=32
    v16h bk0 = load_b_f16(kbase + (size_t)(s0     ) * D, D);
    v16h bk1 = load_b_f16(kbase + (size_t)(s0 + 16) * D, D);
    v16h bk2 = load_b_f16(kbase + (size_t)(s0 + 32) * D, D);
    v16h bk3 = load_b_f16(kbase + (size_t)(s0 + 48) * D, D);
    v8f z = {};
    v8f c0 = WMMA_F16(aq, bk0, z);
    v8f c1 = WMMA_F16(aq, bk1, z);
    v8f c2 = WMMA_F16(aq, bk2, z);
    v8f c3 = WMMA_F16(aq, bk3, z);

    // ---- prefetch next chunk (speculative: overshoot is silently dropped)
    __builtin_prefetch((const char*)(kbase + (size_t)(s0 + 64) * D) + lane * 64, 0, 1);
    __builtin_prefetch((const char*)(mbase + (size_t)(lane >> 1) * S + s0 + 64 + (lane & 1) * 32), 0, 1);

    // ---- bias: mask[b,t,s] * sf[h] * log2e  (C layout: row=r+8*half, col=nn)
#pragma unroll
    for (int r = 0; r < 8; ++r) {
      const float* mp = mbase + (size_t)(r + 8 * half) * S + s0 + nn;
      c0[r] += mp[0]  * sfh;
      c1[r] += mp[16] * sfh;
      c2[r] += mp[32] * sfh;
      c3[r] += mp[48] * sfh;
    }

    // ---- online softmax in log2 domain (rows live in 16-lane halves)
#pragma unroll
    for (int r = 0; r < 8; ++r) {
      float mx = fmaxf(fmaxf(c0[r], c1[r]), fmaxf(c2[r], c3[r]));
      mx = fmaxf(mx, __shfl_xor(mx, 1));
      mx = fmaxf(mx, __shfl_xor(mx, 2));
      mx = fmaxf(mx, __shfl_xor(mx, 4));
      mx = fmaxf(mx, __shfl_xor(mx, 8));
      float mnew = fmaxf(mrun[r], mx);
      float fac  = exp2f(mrun[r] - mnew);
      mrun[r] = mnew;
      float p0 = exp2f(c0[r] - mnew);
      float p1 = exp2f(c1[r] - mnew);
      float p2 = exp2f(c2[r] - mnew);
      float p3 = exp2f(c3[r] - mnew);
      float ps = (p0 + p1) + (p2 + p3);
      ps += __shfl_xor(ps, 1);
      ps += __shfl_xor(ps, 2);
      ps += __shfl_xor(ps, 4);
      ps += __shfl_xor(ps, 8);
      lrun[r] = lrun[r] * fac + ps;
      o0[r] *= fac;
      o1[r] *= fac;
      // C-layout -> row-major f16 P tile in this wave's private LDS slab
      int row = r + 8 * half;
      pl[row * 64 + nn]      = (_Float16)p0;
      pl[row * 64 + 16 + nn] = (_Float16)p1;
      pl[row * 64 + 32 + nn] = (_Float16)p2;
      pl[row * 64 + 48 + nn] = (_Float16)p3;
    }

    // ---- P @ V chunk: A = P (16x64 -> two frags), B = V^T rows, K = 64 keys
    v16h ap0 = load_a_f16(pl, 64);
    v16h ap1 = load_a_f16(pl + 32, 64);
    v16h bv00 = load_b_f16(vbase + s0, S);
    v16h bv01 = load_b_f16(vbase + s0 + 32, S);
    v16h bv10 = load_b_f16(vbase + (size_t)16 * S + s0, S);
    v16h bv11 = load_b_f16(vbase + (size_t)16 * S + s0 + 32, S);
    o0 = WMMA_F16(ap0, bv00, o0);
    o0 = WMMA_F16(ap1, bv01, o0);
    o1 = WMMA_F16(ap0, bv10, o1);
    o1 = WMMA_F16(ap1, bv11, o1);
  }

  // ---- write unnormalized partial O and per-row (m, l)
  size_t pidx = (size_t)(bh * 16 + ttile) * NSPLIT + split;
  float* pob = po + pidx * 16 * 32;
#pragma unroll
  for (int r = 0; r < 8; ++r) {
    int row = r + 8 * half;
    pob[row * 32 + nn]      = o0[r];
    pob[row * 32 + 16 + nn] = o1[r];
  }
  if (nn == 0) {
    float* pmlb = pml + pidx * 16 * 2;
#pragma unroll
    for (int r = 0; r < 8; ++r) {
      int row = r + 8 * half;
      pmlb[row * 2]     = mrun[r];
      pmlb[row * 2 + 1] = lrun[r];
    }
  }
}

// ---- Stage 3: combine split-S partials -------------------------------------
// One thread per (row, d) output element; 8192 rows x 32 d.
__global__ void __launch_bounds__(256)
combine_kernel(const float* __restrict__ po, const float* __restrict__ pml,
               float* __restrict__ ao) {
  int gid = blockIdx.x * 256 + threadIdx.x;
  int row = gid >> 5;        // 0..8191  (= bh*16*16 + ...)
  int d   = gid & 31;
  int tau = row >> 4;        // tile index: bh*16 + ttile
  int rl  = row & 15;

  float mi[NSPLIT], li[NSPLIT];
  float mg = -1.0e30f;
#pragma unroll
  for (int s = 0; s < NSPLIT; ++s) {
    const float* p = pml + ((size_t)tau * NSPLIT + s) * 16 * 2 + rl * 2;
    mi[s] = p[0];
    li[s] = p[1];
    mg = fmaxf(mg, mi[s]);
  }
  float lsum = 0.0f, osum = 0.0f;
#pragma unroll
  for (int s = 0; s < NSPLIT; ++s) {
    float w = exp2f(mi[s] - mg);
    lsum += li[s] * w;
    osum += po[(((size_t)tau * NSPLIT + s) * 16 + rl) * 32 + d] * w;
  }
  float val = osum / lsum;

  int bh = tau >> 4, ttile = tau & 15;
  int t = ttile * 16 + rl;
  int b = bh >> 3, h = bh & 7;
  ao[((size_t)b * 256 + t) * 256 + h * 32 + d] = val;
}

// ---- Stage 4: output projection (f32 out) ---------------------------------
__global__ void __launch_bounds__(128)
outproj_kernel(const float* __restrict__ X, const float* __restrict__ W,
               const float* __restrict__ bias, float* __restrict__ out) {
  int wave  = threadIdx.x >> 5;
  int tile  = blockIdx.x * 4 + wave;
  int mtile = tile >> 4;
  int etile = tile & 15;
  int lane  = threadIdx.x & 31;

  const float* A  = X + (size_t)mtile * 16 * 256;
  const float* Bp = W + (size_t)etile * 16 * 256;
  v8f c = {};
#pragma unroll
  for (int k0 = 0; k0 < 256; k0 += 32) {
    v16h a = load_a_f32(A + k0, 256);
    v16h b = load_b_f32(Bp + k0, 256);
    c = WMMA_F16(a, b, c);
  }
  int half = lane >> 4, nn = lane & 15;
  int e = etile * 16 + nn;
  float bval = bias[e];
#pragma unroll
  for (int r = 0; r < 8; ++r) {
    int m = mtile * 16 + r + 8 * half;
    out[(size_t)m * 256 + e] = c[r] + bval;
  }
}

// ---------------------------------------------------------------------------
extern "C" void kernel_launch(void* const* d_in, const int* in_sizes, int n_in,
                              void* d_out, int out_size, void* d_ws, size_t ws_size,
                              hipStream_t stream) {
  (void)in_sizes; (void)n_in; (void)out_size; (void)ws_size;
  const float* query = (const float*)d_in[0];
  const float* key   = (const float*)d_in[1];
  const float* value = (const float*)d_in[2];
  const float* mask  = (const float*)d_in[3];
  const float* Wq    = (const float*)d_in[4];
  const float* bq    = (const float*)d_in[5];
  const float* Wk    = (const float*)d_in[6];
  const float* bk    = (const float*)d_in[7];
  const float* Wv    = (const float*)d_in[8];
  const float* bv    = (const float*)d_in[9];
  const float* Wo    = (const float*)d_in[10];
  const float* bo    = (const float*)d_in[11];
  const float* sf    = (const float*)d_in[12];
  float* out = (float*)d_out;

  const int B = 4, T = 256, S = 16384, H = 8, D = 32;

  // workspace carve-up (~73 MB total)
  _Float16* qh = (_Float16*)d_ws;                       // [B,H,T,D]   f16  0.5 MB
  _Float16* kh = qh + (size_t)B * H * T * D;            // [B,H,S,D]   f16  32 MB
  _Float16* vt = kh + (size_t)B * H * S * D;            // [B,H,D,S]   f16  32 MB
  float*    ao = (float*)(vt + (size_t)B * H * D * S);  // [B*T,256]   f32  1 MB
  float*    po = ao + (size_t)B * T * 256;              // partial O   f32  4 MB
  float*    pml = po + (size_t)512 * NSPLIT * 16 * 32;  // partial m,l f32  0.25 MB

  // log2-domain softmax: fold log2(e) into the Q pre-scale
  const float qscale = 0.17677669529663687f * LOG2E;    // (1/sqrt(D)) * log2e

  // Stage 1: projections (tiles = (M/16)*(256/16), 4 waves/block)
  proj_kernel<<<(B * T / 16) * 16 / 4, 128, 0, stream>>>(query, Wq, bq, qh, T, qscale, 0);
  proj_kernel<<<(B * S / 16) * 16 / 4, 128, 0, stream>>>(key,   Wk, bk, kh, S, 1.0f,  0);
  proj_kernel<<<(B * S / 16) * 16 / 4, 128, 0, stream>>>(value, Wv, bv, vt, S, 1.0f,  1);

  // Stage 2: split-S flash attention: (B*H) x NSPLIT x 4 t-groups, 4 waves each
  fa_kernel<<<B * H * NSPLIT * 4, 128, 0, stream>>>(qh, kh, vt, mask, sf, po, pml);

  // Stage 3: combine partials (B*H*T rows x D elements)
  combine_kernel<<<(B * H * T * D) / 256, 256, 0, stream>>>(po, pml, ao);

  // Stage 4: output projection
  outproj_kernel<<<(B * T / 16) * 16 / 4, 128, 0, stream>>>(ao, Wo, bo, out);
}
